// LaslNet_15702400434343
// MI455X (gfx1250) — compile-verified, matrix operand order checked by reading, not run on hardware
//
#include <hip/hip_runtime.h>
#include <hip/hip_bf16.h>

// ---------------- problem constants ----------------
#define BATCH 512
#define D1 11
#define D2 9
#define D3 15
#define D4 39
#define O1 9
#define O2 7
#define O3 13
#define O4 37
#define TAPS 81             // 3^4 taps per part
#define KSTEPS 6            // K padded: 6*16 tap slots/part, 81 used
#define NT4 3               // ceil(37/16) tiles along o4
#define TILES_PER_BLK (O3 * NT4)              // 39
#define TILES_PER_B (O1 * O2 * TILES_PER_BLK) // 2457
#define NBLK (BATCH * O1 * O2)                // 32256

// LDS slab: [part(2)][di1(3)][di2(3)][i3(15)][i4(39)] in f16
#define PLANE_HALFS (D3 * D4)               // 585 (contiguous in global AND in slab)
#define NPLANES 18                          // 2*3*3
#define PART_STRIDE (9 * PLANE_HALFS)       // 5265 halves
#define SLAB_HALFS (NPLANES * PLANE_HALFS)  // 10530 -> 21060 bytes LDS

typedef __attribute__((ext_vector_type(16))) _Float16 v16h;
typedef __attribute__((ext_vector_type(8)))  float    v8f;

// Logical K mapping (A and B agree; any bijection onto hardware-K slots is valid):
//   part = lane-group g (runtime, address stride PART_STRIDE)
//   tap  = kstep*16 + e (compile-time constant per unrolled element)
// Tap decodes row-major like the weight tensor: tap = k1*27 + k2*9 + k3*3 + k4.

// ---------------- kernel 1: pack weight A-fragment ----------------
// afrag[(kstep*32 + lane)*16 + e], f16. Rows M = lane%16 (8 used), part = lane/16.
__global__ void pack_w_kernel(const float* __restrict__ w_re,
                              const float* __restrict__ w_im,
                              _Float16* __restrict__ afrag) {
    int t = blockIdx.x * blockDim.x + threadIdx.x;
    if (t >= KSTEPS * 32) return;
    int ks   = t >> 5;
    int lane = t & 31;
    int part = lane >> 4;      // g <-> re/im part
    int m    = lane & 15;
    for (int e = 0; e < 16; ++e) {
        int tap = ks * 16 + e;
        float v = 0.0f;
        if (tap < TAPS && m < 8) {
            int ch = m & 3;
            if (m < 4) {   // real output row: xr*wr - xi*wi
                v = part ? -w_im[ch * TAPS + tap] : w_re[ch * TAPS + tap];
            } else {       // imag output row: xr*wi + xi*wr
                v = part ?  w_re[ch * TAPS + tap] : w_im[ch * TAPS + tap];
            }
        }
        afrag[(ks * 32 + lane) * 16 + e] = (_Float16)v;
    }
}

// ---------------- kernel 2: LDS-staged WMMA conv + fused epilogue ----------------
__global__ __launch_bounds__(256) void conv_wmma_kernel(
    const float* __restrict__ x_re, const float* __restrict__ x_im,
    const _Float16* __restrict__ afrag,
    const float* __restrict__ b_re, const float* __restrict__ b_im,
    const float* __restrict__ lw,
    float* __restrict__ partials) {

    __shared__ _Float16 slab[SLAB_HALFS];

    // block = (b*O1 + o1)*O2 + o2
    int blk = blockIdx.x;
    int o2 = blk % O2;  int r = blk / O2;
    int o1 = r % O1;    int b = r / O1;

    // ---- stage: 18 plane copies; each plane is 585 CONTIGUOUS f32 in global
    //      and 585 contiguous f16 in the slab. Linear inner loop: no div/mod,
    //      coalesced b32 loads -> cvt -> ds_store_b16.
    {
        int pbase = 0;
        for (int p = 0; p < NPLANES; ++p) {       // uniform, SALU-only decode
            int di2 = p % 3;  int q = p / 3;
            int di1 = q % 3;  int part = q / 3;
            const float* src = (part ? x_im : x_re)
                + ((((long)b * D1 + (o1 + di1)) * D2 + (o2 + di2)) * D3) * D4;
            for (int t = threadIdx.x; t < PLANE_HALFS; t += 256)
                slab[pbase + t] = (_Float16)src[t];
            pbase += PLANE_HALFS;
        }
    }
    __syncthreads();

    int wave = threadIdx.x >> 5;   // 8 waves per block (wave32)
    int lane = threadIdx.x & 31;
    int g = lane >> 4;             // K half <-> part
    int n = lane & 15;             // B/C column

    // weights live in registers across all tiles of this slab
    v16h a[KSTEPS];
#pragma unroll
    for (int ks = 0; ks < KSTEPS; ++ks)
        a[ks] = *(const v16h*)(afrag + (ks * 32 + lane) * 16);

    // bias folded into C accumulator: row M = r + 8*g
    float bias[8];
#pragma unroll
    for (int rr = 0; rr < 8; ++rr)
        bias[rr] = (g == 0) ? ((rr < 4) ? b_re[rr] : b_im[rr - 4]) : 0.0f;

    // ---- tile loop: tl = o3*NT4 + t4, waves strided ----
    for (int tl = wave; tl < TILES_PER_BLK; tl += 8) {
        int t4 = tl % NT4;
        int o3 = tl / NT4;
        int ou = t4 * 16 + n;
        bool colv = (ou < O4);
        // runtime part of the gather address (halves); constants folded per element
        int rt = g * PART_STRIDE + o3 * D4 + ou;

        v8f c;
#pragma unroll
        for (int rr = 0; rr < 8; ++rr) c[rr] = bias[rr];

#pragma unroll
        for (int ks = 0; ks < KSTEPS; ++ks) {
            v16h bf = {};
            if (colv) {
#pragma unroll
                for (int e = 0; e < 16; ++e) {
                    const int tap = ks * 16 + e;          // compile-time
                    if (tap < TAPS) {
                        const int k1 = tap / 27;
                        const int k2 = (tap / 9) % 3;
                        const int k3 = (tap / 3) % 3;
                        const int k4 = tap % 3;
                        const int C0 = ((k1 * 3 + k2) * D3 + k3) * D4 + k4; // <= 4760 halves
                        bf[e] = slab[C0 + rt];            // ds_load, imm offset
                    }
                }
            }
            c = __builtin_amdgcn_wmma_f32_16x16x32_f16(
                    /*neg_a=*/false, a[ks], /*neg_b=*/false, bf,
                    /*c_mod=*/(short)0, c, /*reuse_a=*/false, /*reuse_b=*/false);
        }

        // epilogue: lanes 0..15 hold rows M=0..7 in c[0..7]
        float contrib = 0.0f;
        if (g == 0 && colv) {
#pragma unroll
            for (int ch = 0; ch < 4; ++ch) {
                float re = fmaxf(c[ch], 0.0f);        // CReLU (bias already in C)
                float im = fmaxf(c[4 + ch], 0.0f);
                float s = re * re + im * im;
                float mag = (s > 0.0f) ? sqrtf(s) : 0.0f;   // safe sqrt
                long fidx = (((((long)ch * O1 + o1) * O2 + o2) * O3 + o3) * O4) + ou;
                contrib = fmaf(mag, lw[fidx], contrib);
            }
        }

#pragma unroll
        for (int off = 16; off > 0; off >>= 1)
            contrib += __shfl_xor(contrib, off, 32);

        if (lane == 0)
            partials[(long)blk * TILES_PER_BLK + tl] = contrib; // one writer/slot
    }
}

// ---------------- kernel 3: per-sample reduction + sigmoid ----------------
__global__ void finalize_kernel(const float* __restrict__ partials,
                                const float* __restrict__ lb,
                                float* __restrict__ out) {
    int b = blockIdx.x;
    float s = 0.0f;
    for (int i = threadIdx.x; i < TILES_PER_B; i += blockDim.x)
        s += partials[(long)b * TILES_PER_B + i];

    __shared__ float red[256];
    red[threadIdx.x] = s;
    __syncthreads();
    for (int st = 128; st > 0; st >>= 1) {
        if (threadIdx.x < st) red[threadIdx.x] += red[threadIdx.x + st];
        __syncthreads();
    }
    if (threadIdx.x == 0) {
        float z = red[0] + lb[0];
        out[b] = 1.0f / (1.0f + __expf(-z));
    }
}

// ---------------- launcher ----------------
extern "C" void kernel_launch(void* const* d_in, const int* in_sizes, int n_in,
                              void* d_out, int out_size, void* d_ws, size_t ws_size,
                              hipStream_t stream) {
    const float* x_re = (const float*)d_in[0];
    const float* x_im = (const float*)d_in[1];
    const float* w_re = (const float*)d_in[2];
    const float* w_im = (const float*)d_in[3];
    const float* b_re = (const float*)d_in[4];
    const float* b_im = (const float*)d_in[5];
    const float* lw   = (const float*)d_in[6];
    const float* lb   = (const float*)d_in[7];

    // workspace layout:
    //   [0, 6144)        : packed A fragment (f16)
    //   [8192, 8192+~5MB): per-tile partials (f32), NBLK*39 entries
    _Float16* afrag    = (_Float16*)d_ws;
    float*    partials = (float*)((char*)d_ws + 8192);

    pack_w_kernel<<<1, KSTEPS * 32, 0, stream>>>(w_re, w_im, afrag);

    conv_wmma_kernel<<<NBLK, 256, 0, stream>>>(x_re, x_im, afrag,
                                               b_re, b_im, lw, partials);

    finalize_kernel<<<BATCH, 256, 0, stream>>>(partials, lb, (float*)d_out);
}
